// Net_76682346102802
// MI455X (gfx1250) — compile-verified, matrix operand order checked by reading, not run on hardware
//
#include <hip/hip_runtime.h>
#include <hip/hip_bf16.h>

typedef __attribute__((ext_vector_type(16))) _Float16 v16h;
typedef __attribute__((ext_vector_type(8)))  float    v8f;
typedef __attribute__((ext_vector_type(4)))  unsigned v4u;
typedef __attribute__((ext_vector_type(8)))  int      v8i;
typedef __attribute__((ext_vector_type(4)))  int      v4i;

#define N_NODES   131072
#define N_EDGES_C 2097152
#define E_TOT     (N_EDGES_C + N_NODES)
#define HID       32
#define IN_CH     128

#if defined(__HIP_DEVICE_COMPILE__) && __has_builtin(__builtin_amdgcn_tensor_load_to_lds)
#define HAVE_TDM 1
#else
#define HAVE_TDM 0
#endif

__device__ __forceinline__ float leakyf(float v, float s) { return v >= 0.f ? v : s * v; }

// ---- ordered-int encoding for float atomicMax (handles negatives) ----
__device__ __forceinline__ unsigned f32_ord(float f) {
  unsigned u = __float_as_uint(f);
  return (u & 0x80000000u) ? ~u : (u | 0x80000000u);
}
__device__ __forceinline__ float ord_f32(unsigned u) {
  return __uint_as_float((u & 0x80000000u) ? (u & 0x7fffffffu) : ~u);
}

#if HAVE_TDM
// Tensor Data Mover: 2-D tile (rows x cols f32, row pitch stride_elts) -> LDS.
// D# per CDNA5 ISA: group0 {count=1, lds_addr, global_addr, type=2},
// group1 {data_size=4B, tensor_dim0/1, tile_dim0/1, tensor_dim0_stride}.
__device__ __forceinline__ void tdm_load_2d_f32(unsigned lds_off, const void* gaddr,
                                                unsigned rows, unsigned cols,
                                                unsigned stride_elts) {
  unsigned long long ga = (unsigned long long)(size_t)gaddr;
  v4u g0;
  g0[0] = 1u;                                           // count=1 (user descriptor)
  g0[1] = lds_off;                                      // lds_addr (bytes)
  g0[2] = (unsigned)(ga & 0xffffffffu);                 // global_addr[31:0]
  g0[3] = (unsigned)((ga >> 32) & 0x01ffffffu) | 0x80000000u; // addr[56:32] | type=2<<30
  v8i g1;
  g1[0] = (int)(0x2u << 16);                            // data_size=2 -> 4 bytes
  g1[1] = (int)((cols & 0xffffu) << 16);                // tensor_dim0[15:0] @ bits63:48
  g1[2] = (int)(((cols >> 16) & 0xffffu) | ((rows & 0xffffu) << 16)); // dim0 hi | dim1 lo
  g1[3] = (int)(((rows >> 16) & 0xffffu) | ((cols & 0xffffu) << 16)); // dim1 hi | tile_dim0
  g1[4] = (int)(rows & 0xffffu);                        // tile_dim1 (tile_dim2=0)
  g1[5] = (int)stride_elts;                             // tensor_dim0_stride[31:0]
  g1[6] = 0;
  g1[7] = 0;
  v4i zz = {0, 0, 0, 0};
#if defined(__clang_major__) && (__clang_major__ >= 23)
  v8i z8 = {0, 0, 0, 0, 0, 0, 0, 0};
  __builtin_amdgcn_tensor_load_to_lds(g0, g1, zz, zz, z8, 0);
#else
  __builtin_amdgcn_tensor_load_to_lds(g0, g1, zz, zz, 0);
#endif
}
#endif

// Map packed-fragment element index -> source index in row-major W[k][col] (HID cols).
// Fragment layout: wfrag[ks][nt][lane][e], lane=(hf,idx), e<8 -> K=ks*32+8*hf+e,
// e>=8 -> K=ks*32+16+8*hf+(e-8); col = nt*16+idx.  One v16h per (ks,nt,lane).
__device__ __forceinline__ int frag_src_index(int i) {
  int e    = i & 15;
  int lane = (i >> 4) & 31;
  int nt   = (i >> 9) & 1;
  int ks   = i >> 10;
  int hf = lane >> 4, idx = lane & 15;
  int k = ks * 32 + 8 * hf + (e < 8 ? e : e + 8);
  return k * HID + nt * 16 + idx;
}

// =====================================================================
// Fill kernels
// =====================================================================
__global__ void fill_f32(float* __restrict__ p, float v, int n) {
  int i = blockIdx.x * blockDim.x + threadIdx.x;
  if (i < n) p[i] = v;
}
__global__ void fill_u32(unsigned* __restrict__ p, unsigned v, int n) {
  int i = blockIdx.x * blockDim.x + threadIdx.x;
  if (i < n) p[i] = v;
}

// =====================================================================
// Embed: h = x @ w_embed + b_embed   (131072x128 @ 128x32)
// One wave -> 16 rows x 32 cols; K=128 in 4 WMMA steps of 32.
// W staged global->LDS by TDM, then swizzled into fragment order so each
// B fragment is one contiguous 32-byte LDS read.
// =====================================================================
__global__ void embed_kernel(const float* __restrict__ x,
                             const float* __restrict__ w,
                             const float* __restrict__ bias,
                             float* __restrict__ h) {
  __shared__ __align__(16) float    wraw[IN_CH * HID];
  __shared__ __align__(32) _Float16 wfrag[4 * 2 * 32 * 16];
  int tid = threadIdx.x;
#if HAVE_TDM
  if (tid < 32) {
    if (tid == 0) {
      tdm_load_2d_f32((unsigned)(size_t)&wraw[0], w, IN_CH, HID, HID);
    }
    __builtin_amdgcn_s_wait_tensorcnt(0);
  }
#else
  for (int i = tid; i < IN_CH * HID; i += 256) wraw[i] = w[i];
#endif
  __syncthreads();
  for (int i = tid; i < 4096; i += 256) wfrag[i] = (_Float16)wraw[frag_src_index(i)];
  __syncthreads();

  int wave = tid >> 5, lane = tid & 31;
  int hf = lane >> 4, idx = lane & 15;
  int rowbase = blockIdx.x * 128 + wave * 16;
  int row = rowbase + idx;
  const v16h* bfrag = (const v16h*)wfrag;

  v8f c0 = {}, c1 = {};
  for (int ks = 0; ks < 4; ++ks) {
    const float* xr = x + (size_t)row * IN_CH + ks * 32 + 8 * hf;
    if (ks < 3) __builtin_prefetch(xr + 32, 0, 0);
    v16h a;
#pragma unroll
    for (int e = 0; e < 8; ++e) { a[e] = (_Float16)xr[e]; a[8 + e] = (_Float16)xr[16 + e]; }
    v16h b0 = bfrag[(ks * 2 + 0) * 32 + lane];
    v16h b1 = bfrag[(ks * 2 + 1) * 32 + lane];
    c0 = __builtin_amdgcn_wmma_f32_16x16x32_f16(false, a, false, b0, (short)0, c0, false, false);
    c1 = __builtin_amdgcn_wmma_f32_16x16x32_f16(false, a, false, b1, (short)0, c1, false, false);
  }
#pragma unroll
  for (int r = 0; r < 8; ++r) {
    int m = rowbase + r + 8 * hf;
    h[(size_t)m * HID + idx]      = c0[r] + bias[idx];
    h[(size_t)m * HID + 16 + idx] = c1[r] + bias[16 + idx];
  }
}

// =====================================================================
// BatchNorm (eval) + leaky 0.01
// =====================================================================
__global__ void bn_leaky_kernel(const float* __restrict__ h,
                                const float* __restrict__ m, const float* __restrict__ v,
                                const float* __restrict__ g, const float* __restrict__ b,
                                float* __restrict__ t) {
  int i = blockIdx.x * blockDim.x + threadIdx.x;
  if (i >= N_NODES * HID) return;
  int c = i & 31;
  float val = (h[i] - m[c]) * rsqrtf(v[c] + 1e-5f) * g[c] + b[c];
  t[i] = leakyf(val, 0.01f);
}

// =====================================================================
// GAT GEMM: ht = t @ W (32x32). One wave -> 16x32 tile, single K step.
// =====================================================================
__global__ void gat_gemm_kernel(const float* __restrict__ t,
                                const float* __restrict__ w,
                                float* __restrict__ ht) {
  __shared__ __align__(16) float    wraw[HID * HID];
  __shared__ __align__(32) _Float16 wfrag[2 * 32 * 16];
  int tid = threadIdx.x;
#if HAVE_TDM
  if (tid < 32) {
    if (tid == 0) {
      tdm_load_2d_f32((unsigned)(size_t)&wraw[0], w, HID, HID, HID);
    }
    __builtin_amdgcn_s_wait_tensorcnt(0);
  }
#else
  for (int i = tid; i < HID * HID; i += 256) wraw[i] = w[i];
#endif
  __syncthreads();
  for (int i = tid; i < 1024; i += 256) wfrag[i] = (_Float16)wraw[frag_src_index(i)];
  __syncthreads();

  int wave = tid >> 5, lane = tid & 31;
  int hf = lane >> 4, idx = lane & 15;
  int rowbase = blockIdx.x * 128 + wave * 16;
  int row = rowbase + idx;
  const v16h* bfrag = (const v16h*)wfrag;

  const float* xr = t + (size_t)row * HID + 8 * hf;
  v16h a;
#pragma unroll
  for (int e = 0; e < 8; ++e) { a[e] = (_Float16)xr[e]; a[8 + e] = (_Float16)xr[16 + e]; }
  v16h b0 = bfrag[lane];
  v16h b1 = bfrag[32 + lane];
  v8f c0 = {}, c1 = {};
  c0 = __builtin_amdgcn_wmma_f32_16x16x32_f16(false, a, false, b0, (short)0, c0, false, false);
  c1 = __builtin_amdgcn_wmma_f32_16x16x32_f16(false, a, false, b1, (short)0, c1, false, false);
#pragma unroll
  for (int r = 0; r < 8; ++r) {
    int m = rowbase + r + 8 * hf;
    ht[(size_t)m * HID + idx]      = c0[r];
    ht[(size_t)m * HID + 16 + idx] = c1[r];
  }
}

// =====================================================================
// Per-node attention dots: alpha_src = ht . a_src, alpha_dst = ht . a_dst
// =====================================================================
__global__ void alpha_kernel(const float* __restrict__ ht,
                             const float* __restrict__ asv, const float* __restrict__ adv,
                             float* __restrict__ as_out, float* __restrict__ ad_out) {
  int node = blockIdx.x * blockDim.x + threadIdx.x;
  if (node >= N_NODES) return;
  const float* hr = ht + (size_t)node * HID;
  float s = 0.f, d = 0.f;
#pragma unroll
  for (int f = 0; f < HID; ++f) { s += hr[f] * asv[f]; d += hr[f] * adv[f]; }
  as_out[node] = s;
  ad_out[node] = d;
}

// =====================================================================
// Edge pass 1: logit + segment max over dst (ordered-int atomicMax)
// =====================================================================
__global__ void edge_logit_max_kernel(const int* __restrict__ src, const int* __restrict__ dst,
                                      const float* __restrict__ as, const float* __restrict__ ad,
                                      float* __restrict__ logit, unsigned* __restrict__ menc) {
  int e = blockIdx.x * blockDim.x + threadIdx.x;
  if (e >= E_TOT) return;
  int s = (e < N_EDGES_C) ? src[e] : (e - N_EDGES_C);
  int d = (e < N_EDGES_C) ? dst[e] : (e - N_EDGES_C);
  float l = leakyf(as[s] + ad[d], 0.2f);
  logit[e] = l;
  atomicMax(&menc[d], f32_ord(l));
}

__global__ void m_decode_kernel(const unsigned* __restrict__ menc, float* __restrict__ mf) {
  int i = blockIdx.x * blockDim.x + threadIdx.x;
  if (i >= N_NODES) return;
  float v = ord_f32(menc[i]);
  if (!__builtin_isfinite(v)) v = 0.f;
  mf[i] = v;
}

// =====================================================================
// Edge pass 2: e = exp(logit - m[dst]); z[dst] += e
// =====================================================================
__global__ void edge_exp_sum_kernel(const int* __restrict__ dst,
                                    const float* __restrict__ logit, const float* __restrict__ mf,
                                    float* __restrict__ ev, float* __restrict__ z) {
  int e = blockIdx.x * blockDim.x + threadIdx.x;
  if (e >= E_TOT) return;
  int d = (e < N_EDGES_C) ? dst[e] : (e - N_EDGES_C);
  float v = __expf(logit[e] - mf[d]);
  ev[e] = v;
  atomicAdd(&z[d], v);
}

// =====================================================================
// Edge pass 3: agg[dst] += (e / (z[dst]+eps)) * ht[src]; 8 threads/edge, 4ch each
// =====================================================================
__global__ void edge_scatter_kernel(const int* __restrict__ src, const int* __restrict__ dst,
                                    const float* __restrict__ ev, const float* __restrict__ z,
                                    const float* __restrict__ ht, float* __restrict__ agg) {
  long long g = (long long)blockIdx.x * blockDim.x + threadIdx.x;
  int e = (int)(g >> 3);
  if (e >= E_TOT) return;
  int c0 = ((int)g & 7) * 4;
  int s = (e < N_EDGES_C) ? src[e] : (e - N_EDGES_C);
  int d = (e < N_EDGES_C) ? dst[e] : (e - N_EDGES_C);
  float alpha = ev[e] / (z[d] + 1e-16f);
  const float4 hv = *(const float4*)(ht + (size_t)s * HID + c0);
  float* ag = agg + (size_t)d * HID + c0;
  atomicAdd(&ag[0], alpha * hv.x);
  atomicAdd(&ag[1], alpha * hv.y);
  atomicAdd(&ag[2], alpha * hv.z);
  atomicAdd(&ag[3], alpha * hv.w);
}

// =====================================================================
// Residual: h += agg + gat_b
// =====================================================================
__global__ void residual_kernel(float* __restrict__ h, const float* __restrict__ agg,
                                const float* __restrict__ gb) {
  int i = blockIdx.x * blockDim.x + threadIdx.x;
  if (i >= N_NODES * HID) return;
  h[i] = h[i] + agg[i] + gb[i & 31];
}

// =====================================================================
// MemPool1: student-t kernel, head-norm, 1x1 conv over heads, softmax.
// Writes S1 [N,10], accumulates xo [64,10,32] and colsum [64,10] via LDS.
// One block = 256 consecutive nodes (same graph; 256 | 2048).
// =====================================================================
__global__ void mempool1_kernel(const float* __restrict__ h,
                                const float* __restrict__ k1, const float* __restrict__ convw,
                                float* __restrict__ S1, float* __restrict__ xo,
                                float* __restrict__ colsum) {
  __shared__ float keys[5 * 10 * HID];
  __shared__ float knorm[50];
  __shared__ float xo_s[10 * HID];
  __shared__ float cs_s[10];
  int tid = threadIdx.x;
  for (int i = tid; i < 5 * 10 * HID; i += 256) keys[i] = k1[i];
  for (int i = tid; i < 10 * HID; i += 256) xo_s[i] = 0.f;
  if (tid < 10) cs_s[tid] = 0.f;
  __syncthreads();
  if (tid < 50) {
    const float* kr = keys + tid * HID;
    float s = 0.f;
    for (int f = 0; f < HID; ++f) s += kr[f] * kr[f];
    knorm[tid] = s;
  }
  __syncthreads();

  int node = blockIdx.x * 256 + tid;
  const float* hp = h + (size_t)node * HID;
  float xr[HID];
  float x2 = 0.f;
#pragma unroll
  for (int f = 0; f < HID; ++f) { xr[f] = hp[f]; x2 += xr[f] * xr[f]; }

  float Sc[10];
  for (int k = 0; k < 10; ++k) Sc[k] = 0.f;
  for (int hh = 0; hh < 5; ++hh) {
    float Sh[10], rs = 0.f;
    for (int k = 0; k < 10; ++k) {
      const float* kr = keys + (hh * 10 + k) * HID;
      float dot = 0.f;
#pragma unroll
      for (int f = 0; f < HID; ++f) dot += xr[f] * kr[f];
      float dd = fmaxf(x2 + knorm[hh * 10 + k] - 2.f * dot, 0.f);
      float s = 1.f / (1.f + dd);   // (1+d)^(-(tau+1)/2), tau=1
      Sh[k] = s; rs += s;
    }
    float inv = 1.f / rs;
    float cw = convw[hh];
    for (int k = 0; k < 10; ++k) Sc[k] += cw * Sh[k] * inv;
  }
  // softmax over clusters
  float mx = Sc[0];
  for (int k = 1; k < 10; ++k) mx = fmaxf(mx, Sc[k]);
  float ssum = 0.f, Sr[10];
  for (int k = 0; k < 10; ++k) { Sr[k] = __expf(Sc[k] - mx); ssum += Sr[k]; }
  float inv = 1.f / ssum;
  for (int k = 0; k < 10; ++k) {
    Sr[k] *= inv;
    S1[(size_t)node * 10 + k] = Sr[k];
    atomicAdd(&cs_s[k], Sr[k]);
  }
  for (int k = 0; k < 10; ++k)
    for (int f = 0; f < HID; ++f)
      atomicAdd(&xo_s[k * HID + f], Sr[k] * xr[f]);
  __syncthreads();
  int gb = blockIdx.x >> 3;  // graph id (8 blocks per graph)
  for (int i = tid; i < 10 * HID; i += 256) atomicAdd(&xo[gb * 10 * HID + i], xo_s[i]);
  if (tid < 10) atomicAdd(&colsum[gb * 10 + tid], cs_s[tid]);
}

// x1[b,k,j] = xo[b,k,:] @ lin1_w[:,j] + lin1_b[j]
__global__ void mp1_fin_kernel(const float* __restrict__ xo, const float* __restrict__ w1,
                               const float* __restrict__ b1, float* __restrict__ x1) {
  int idx = blockIdx.x * blockDim.x + threadIdx.x;
  if (idx >= 64 * 10 * 80) return;
  int j = idx % 80;
  int bk = idx / 80;
  const float* xr = xo + bk * HID;
  float acc = b1[j];
#pragma unroll
  for (int f = 0; f < HID; ++f) acc += xr[f] * w1[f * 80 + j];
  x1[idx] = acc;
}

// kl_loss(S1): P = S^2/colsum; denom = sum_k P; loss = sum P*(lnP - lnS)
__global__ void kl1_kernel(const float* __restrict__ S1, const float* __restrict__ colsum,
                           float* __restrict__ kl_acc) {
  __shared__ float red[256];
  int tid = threadIdx.x;
  int node = blockIdx.x * 256 + tid;
  int b = node >> 11;
  float srow[10], ssum = 0.f;
  for (int k = 0; k < 10; ++k) { srow[k] = S1[(size_t)node * 10 + k]; ssum += srow[k]; }
  float P[10], denom = 0.f;
  for (int k = 0; k < 10; ++k) {
    P[k] = srow[k] * srow[k] / fmaxf(colsum[b * 10 + k], 1e-15f);
    denom += P[k];
  }
  if (ssum == 0.f) denom = 1.f;
  float contrib = 0.f;
  for (int k = 0; k < 10; ++k) {
    float p = fmaxf(P[k] / denom, 1e-15f);
    contrib += p * (__logf(p) - __logf(fmaxf(srow[k], 1e-15f)));
  }
  red[tid] = contrib;
  __syncthreads();
  for (int s = 128; s > 0; s >>= 1) { if (tid < s) red[tid] += red[tid + s]; __syncthreads(); }
  if (tid == 0) atomicAdd(kl_acc, red[0]);
}

// =====================================================================
// MemPool2 (K=1 => S2 == 1 after softmax) + lin2 + leaky + log_softmax
// + kl(S2) + final kl sum. One block, 64 threads (one per graph).
// =====================================================================
__global__ void final_kernel(const float* __restrict__ x1,
                             const float* __restrict__ k2, const float* __restrict__ cw2,
                             const float* __restrict__ w2, const float* __restrict__ b2,
                             const float* __restrict__ kl1_acc, float* __restrict__ out) {
  __shared__ float klred[64];
  int b = threadIdx.x;
  float S2v[10];
  float cs2 = 0.f;
  for (int n = 0; n < 10; ++n) {
    // faithful student-t over 5 heads, 1 cluster; head-norm over K=1 gives 1,
    // conv gives a per-node-constant logit, softmax over K=1 gives exactly 1.
    const float* xrp = x1 + ((size_t)b * 10 + n) * 80;
    float xn2 = 0.f;
    for (int f = 0; f < 80; ++f) { float v = leakyf(xrp[f], 0.01f); xn2 += v * v; }
    float sc = 0.f;
    for (int hh = 0; hh < 5; ++hh) {
      const float* kr = k2 + hh * 80;
      float kn = 0.f, dot = 0.f;
      for (int f = 0; f < 80; ++f) {
        float v = leakyf(xrp[f], 0.01f);
        kn += kr[f] * kr[f]; dot += v * kr[f];
      }
      float dd = fmaxf(xn2 + kn - 2.f * dot, 0.f);
      float s = 1.f / (1.f + dd);
      sc += cw2[hh] * (s / s);
    }
    float sm = __expf(sc - sc);  // softmax over single cluster -> 1
    S2v[n] = sm;
    cs2 += sm;
  }
  // xo2[f] = sum_n S2*leaky(x1); y = xo2 @ w2 + b2
  float y[10];
  for (int j = 0; j < 10; ++j) y[j] = b2[j];
  for (int f = 0; f < 80; ++f) {
    float acc = 0.f;
    for (int n = 0; n < 10; ++n)
      acc += S2v[n] * leakyf(x1[((size_t)b * 10 + n) * 80 + f], 0.01f);
    for (int j = 0; j < 10; ++j) y[j] += acc * w2[f * 10 + j];
  }
  float mx = -1e30f;
  for (int j = 0; j < 10; ++j) { y[j] = leakyf(y[j], 0.01f); mx = fmaxf(mx, y[j]); }
  float se = 0.f;
  for (int j = 0; j < 10; ++j) se += __expf(y[j] - mx);
  float lse = mx + __logf(se);
  for (int j = 0; j < 10; ++j) out[b * 10 + j] = y[j] - lse;
  // kl for S2 [64,10,1]
  float kl = 0.f;
  for (int n = 0; n < 10; ++n) {
    float s = S2v[n];
    float P = s * s / fmaxf(cs2, 1e-15f);
    float denom = P;
    if (s == 0.f) denom = 1.f;
    float p = fmaxf(P / denom, 1e-15f);
    kl += p * (__logf(p) - __logf(fmaxf(s, 1e-15f)));
  }
  klred[b] = kl;
  __syncthreads();
  for (int st = 32; st > 0; st >>= 1) { if (b < st) klred[b] += klred[b + st]; __syncthreads(); }
  if (b == 0) out[640] = kl1_acc[0] / 64.f + klred[0] / 64.f;
}

// =====================================================================
extern "C" void kernel_launch(void* const* d_in, const int* in_sizes, int n_in,
                              void* d_out, int out_size, void* d_ws, size_t ws_size,
                              hipStream_t stream) {
  const float* x       = (const float*)d_in[0];
  const int*   ei      = (const int*)d_in[1];   // [2, E]: row0=src, row1=dst
  const float* w_embed = (const float*)d_in[3];
  const float* b_embed = (const float*)d_in[4];
  const float* gw[2]  = {(const float*)d_in[5],  (const float*)d_in[13]};
  const float* asr[2] = {(const float*)d_in[6],  (const float*)d_in[14]};
  const float* ads[2] = {(const float*)d_in[7],  (const float*)d_in[15]};
  const float* gbp[2] = {(const float*)d_in[8],  (const float*)d_in[16]};
  const float* bg[2]  = {(const float*)d_in[9],  (const float*)d_in[17]};
  const float* bb[2]  = {(const float*)d_in[10], (const float*)d_in[18]};
  const float* bm[2]  = {(const float*)d_in[11], (const float*)d_in[19]};
  const float* bv[2]  = {(const float*)d_in[12], (const float*)d_in[20]};
  const float* k1      = (const float*)d_in[21];
  const float* conv1w  = (const float*)d_in[22];
  const float* lin1w   = (const float*)d_in[23];
  const float* lin1b   = (const float*)d_in[24];
  const float* k2      = (const float*)d_in[25];
  const float* conv2w  = (const float*)d_in[26];
  const float* lin2w   = (const float*)d_in[27];
  const float* lin2b   = (const float*)d_in[28];

  const int* src = ei;
  const int* dst = ei + N_EDGES_C;

  // workspace layout (floats)
  float* W = (float*)d_ws;
  size_t off = 0;
  float*    h     = W + off; off += (size_t)N_NODES * HID;
  float*    t     = W + off; off += (size_t)N_NODES * HID;
  float*    ht    = W + off; off += (size_t)N_NODES * HID;
  float*    agg   = W + off; off += (size_t)N_NODES * HID;
  float*    a_s   = W + off; off += N_NODES;
  float*    a_d   = W + off; off += N_NODES;
  unsigned* menc  = (unsigned*)(W + off); off += N_NODES;
  float*    mf    = W + off; off += N_NODES;
  float*    z     = W + off; off += N_NODES;
  float*    logit = W + off; off += E_TOT;
  float*    ev    = W + off; off += E_TOT;
  float*    S1    = W + off; off += (size_t)N_NODES * 10;
  float*    xo1   = W + off; off += 64 * 10 * HID;
  float*    x1    = W + off; off += 64 * 10 * 80;
  float*    cs1   = W + off; off += 64 * 10;
  float*    kl1a  = W + off; off += 1;

  const int T = 256;
  const int NB_N   = (N_NODES + T - 1) / T;
  const int NB_NH  = (N_NODES * HID + T - 1) / T;
  const int NB_E   = (E_TOT + T - 1) / T;
  const int NB_E8  = (int)(((long long)E_TOT * 8 + T - 1) / T);

  embed_kernel<<<N_NODES / 128, T, 0, stream>>>(x, w_embed, b_embed, h);

  for (int l = 0; l < 2; ++l) {
    fill_u32<<<NB_N, T, 0, stream>>>(menc, 0x007FFFFFu, N_NODES);  // enc(-inf)
    fill_f32<<<NB_N, T, 0, stream>>>(z, 0.f, N_NODES);
    fill_f32<<<NB_NH, T, 0, stream>>>(agg, 0.f, N_NODES * HID);
    bn_leaky_kernel<<<NB_NH, T, 0, stream>>>(h, bm[l], bv[l], bg[l], bb[l], t);
    gat_gemm_kernel<<<N_NODES / 128, T, 0, stream>>>(t, gw[l], ht);
    alpha_kernel<<<NB_N, T, 0, stream>>>(ht, asr[l], ads[l], a_s, a_d);
    edge_logit_max_kernel<<<NB_E, T, 0, stream>>>(src, dst, a_s, a_d, logit, menc);
    m_decode_kernel<<<NB_N, T, 0, stream>>>(menc, mf);
    edge_exp_sum_kernel<<<NB_E, T, 0, stream>>>(dst, logit, mf, ev, z);
    edge_scatter_kernel<<<NB_E8, T, 0, stream>>>(src, dst, ev, z, ht, agg);
    residual_kernel<<<NB_NH, T, 0, stream>>>(h, agg, gbp[l]);
  }

  fill_f32<<<(64 * 10 * HID + T - 1) / T, T, 0, stream>>>(xo1, 0.f, 64 * 10 * HID);
  fill_f32<<<1, T, 0, stream>>>(cs1, 0.f, 64 * 10);
  fill_f32<<<1, T, 0, stream>>>(kl1a, 0.f, 1);

  mempool1_kernel<<<N_NODES / T, T, 0, stream>>>(h, k1, conv1w, S1, xo1, cs1);
  mp1_fin_kernel<<<(64 * 10 * 80 + T - 1) / T, T, 0, stream>>>(xo1, lin1w, lin1b, x1);
  kl1_kernel<<<N_NODES / T, T, 0, stream>>>(S1, cs1, kl1a);
  final_kernel<<<1, 64, 0, stream>>>(x1, k2, conv2w, lin2w, lin2b, kl1a, (float*)d_out);
}
